// VoxelNeXtBackbone_71012989272460
// MI455X (gfx1250) — matrix operand order repaired and checked by reading, not compiled, not run on hardware
//
#include <hip/hip_runtime.h>

// ---------------------------------------------------------------------------
// VoxelNeXt backbone on MI455X (gfx1250).
// Implicit-GEMM 3D convs: V_WMMA_F32_16X16X32_BF16, fed by CDNA5
// GLOBAL_LOAD_ASYNC_TO_LDS_B128 with ASYNCcnt-based double buffering
// (no VGPR staging, no workgroup barriers: one wave per block).
// Activations: bf16, channels-last, channel stride padded to max(32, C).
// Each wave computes a 16(M) x 16*NT(N) tile (NT=4 for C>=64): one A fragment
// is reused across NT WMMAs per K-chunk.  All indexing is shift/mask math.
// ---------------------------------------------------------------------------

typedef __attribute__((ext_vector_type(16))) __bf16 v16bf;
typedef __attribute__((ext_vector_type(8)))  float  v8f;

__device__ __forceinline__ unsigned short f2bf(float f) {
  union { float f; unsigned u; } x; x.f = f;
  unsigned r = x.u + 0x7fffu + ((x.u >> 16) & 1u);   // round-to-nearest-even
  return (unsigned short)(r >> 16);
}
__device__ __forceinline__ float bf2f(unsigned short h) {
  union { unsigned u; float f; } x; x.u = ((unsigned)h) << 16;
  return x.f;
}

// ---------------------------------------------------------------------------
// Generic conv kernel (template NT = number of 16-wide N sub-tiles per wave).
//  mode 0: stride-1, pad 1           (zi = zo + k - 1)
//  mode 1: stride-2, pad 1           (zi = 2*zo + k - 1)
//  mode 2: transposed stride-2 (lhs_dilation=2, pad(1,2), k=3):
//          j = zo + k - 1 contributes iff j even and 0 <= j/2 < Din
// in : [B, Din^3, 1<<csiS] bf16 (channels-last)
// out: [B, Do ^3, 1<<csoS] bf16
// wb : [27, 1<<csiS, 1<<coutS] bf16
// ---------------------------------------------------------------------------
template <int NT>
__global__ __launch_bounds__(32) void conv_wmma_k(
    const unsigned short* __restrict__ in, unsigned short* __restrict__ out,
    const unsigned short* __restrict__ wb,
    const float* __restrict__ bn_g, const float* __restrict__ bn_b,
    const float* __restrict__ bn_m, const float* __restrict__ bn_v,
    const float* __restrict__ mask, const unsigned short* __restrict__ resid,
    int Din, int dinS, int doS, int csiS, int csoS, int coutS,
    int mode, int relu)
{
  // LDS layout (single object -> base offset 0):
  //   [0      , 2048)           : 2 x A buffer (16 rows x 32 k bf16 = 1KB)
  //   [2048   , 2048+2048*NT)   : 2 x B buffer (32 k x 16*NT n bf16 = 1KB*NT)
  __shared__ __align__(16) unsigned short ldsMem[1024 + 1024 * NT];

  const int lane  = threadIdx.x;
  const int mBase = blockIdx.x * 16;
  const int n0    = blockIdx.y * (16 * NT);

  // Row this lane stages into the A tile: row = lane/2, half = lane&1.
  const int sRow  = lane >> 1;
  const int sHalf = lane & 1;

  const int Do  = 1 << doS;
  const int row = mBase + sRow;
  const int b_  = row >> (3 * doS);
  const int rem = row & ((1 << (3 * doS)) - 1);
  const int zo  = rem >> (2 * doS);
  const int yo  = (rem >> doS) & (Do - 1);
  const int xo  = rem & (Do - 1);

  v8f acc[NT];
  #pragma unroll
  for (int s = 0; s < NT; ++s)
    #pragma unroll
    for (int j = 0; j < 8; ++j) acc[s][j] = 0.0f;

  const int ktS   = csiS - 5;       // log2(K-chunks of 32)
  const int kT    = 1 << ktS;
  const int totIt = 27 << ktS;

  // Fragment-gather constants (wave32 WMMA bf16 layouts).
  const int r   = lane & 15;                 // A row
  const int kb  = (lane & 16) ? 4 : 0;       // A dword offset (= K/2)
  const int nn  = lane & 15;                 // B col within sub-tile
  const int kb2 = (lane & 16) ? 8 : 0;       // B K base

  const unsigned long long inB = (unsigned long long)(size_t)in;
  const unsigned long long wbB = (unsigned long long)(size_t)wb;

  // Issue the async global->LDS loads for iteration `it`; returns whether this
  // lane's A row is valid (invalid lanes load from offset 0 and are zeroed
  // after the wait, keeping the per-tile issue count uniform).
  auto issue = [&](int it) -> bool {
    const int kc = it & (kT - 1);
    const int t  = it >> ktS;
    const int kd = t / 9, rr2 = t - kd * 9, kh = rr2 / 3, kw = rr2 - kh * 3;
    int zi, yi, xi; bool ok;
    if (mode == 0)      { zi = zo + kd - 1;   yi = yo + kh - 1;   xi = xo + kw - 1;   ok = true; }
    else if (mode == 1) { zi = 2*zo + kd - 1; yi = 2*yo + kh - 1; xi = 2*xo + kw - 1; ok = true; }
    else {
      const int jz = zo + kd - 1, jy = yo + kh - 1, jx = xo + kw - 1;
      ok = (((jz | jy | jx) & 1) == 0);       // all dims even (negatives excluded)
      zi = jz >> 1; yi = jy >> 1; xi = jx >> 1;
    }
    ok = ok && (unsigned)zi < (unsigned)Din && (unsigned)yi < (unsigned)Din
            && (unsigned)xi < (unsigned)Din;

    unsigned aOff = 0;   // byte offset from `in`
    if (ok) {
      const unsigned vox =
          (unsigned)((((((b_ << dinS) + zi) << dinS) + yi) << dinS) + xi);
      aOff = ((vox << csiS) + (unsigned)((kc << 5) + sHalf * 16)) * 2u;
    }

    const unsigned ldsA = (unsigned)((it & 1) * 1024 + (sRow * 32 + sHalf * 16) * 2);
    asm volatile(
        "global_load_async_to_lds_b128 %0, %1, %2 offset:0\n\t"
        "global_load_async_to_lds_b128 %0, %1, %2 offset:16"
        :: "v"(ldsA), "v"(aOff), "s"(inB) : "memory");

    const unsigned bOff =
        (unsigned)((((t << csiS) + (kc << 5) + lane) << coutS) + n0) * 2u;
    const unsigned ldsB =
        (unsigned)(2048 + (it & 1) * 1024 * NT + lane * (16 * NT) * 2);
    if (NT == 1) {
      asm volatile(
          "global_load_async_to_lds_b128 %0, %1, %2 offset:0\n\t"
          "global_load_async_to_lds_b128 %0, %1, %2 offset:16"
          :: "v"(ldsB), "v"(bOff), "s"(wbB) : "memory");
    } else if (NT == 2) {
      asm volatile(
          "global_load_async_to_lds_b128 %0, %1, %2 offset:0\n\t"
          "global_load_async_to_lds_b128 %0, %1, %2 offset:16\n\t"
          "global_load_async_to_lds_b128 %0, %1, %2 offset:32\n\t"
          "global_load_async_to_lds_b128 %0, %1, %2 offset:48"
          :: "v"(ldsB), "v"(bOff), "s"(wbB) : "memory");
    } else {
      asm volatile(
          "global_load_async_to_lds_b128 %0, %1, %2 offset:0\n\t"
          "global_load_async_to_lds_b128 %0, %1, %2 offset:16\n\t"
          "global_load_async_to_lds_b128 %0, %1, %2 offset:32\n\t"
          "global_load_async_to_lds_b128 %0, %1, %2 offset:48\n\t"
          "global_load_async_to_lds_b128 %0, %1, %2 offset:64\n\t"
          "global_load_async_to_lds_b128 %0, %1, %2 offset:80\n\t"
          "global_load_async_to_lds_b128 %0, %1, %2 offset:96\n\t"
          "global_load_async_to_lds_b128 %0, %1, %2 offset:112"
          :: "v"(ldsB), "v"(bOff), "s"(wbB) : "memory");
    }
    return ok;
  };

  bool okCur = issue(0);

  for (int it = 0; it < totIt; ++it) {
    bool okNext = true;
    if (it + 1 < totIt) {
      okNext = issue(it + 1);
      // Async loads complete in order and every wave issues the same count per
      // tile, so <= (next tile's loads) outstanding => current tile complete.
      asm volatile("s_wait_asynccnt %0" :: "i"(2 + 2 * NT) : "memory");
    } else {
      asm volatile("s_wait_asynccnt 0x0" ::: "memory");
    }

    unsigned short* bufA = ldsMem + (it & 1) * 512;
    const unsigned short* bufB = ldsMem + 1024 + (it & 1) * 512 * NT;

    // Zero this lane's A row if its tap was out of bounds (LDS ops are
    // in-order per wave, so the fragment ds_loads below observe the zeros).
    if (!okCur) {
      const uint4 z = make_uint4(0u, 0u, 0u, 0u);
      uint4* dA = (uint4*)&bufA[sRow * 32 + sHalf * 16];
      dA[0] = z; dA[1] = z;
    }

    // ---- A fragment (16-bit A 16x32 layout): lanes<16 hold K 0..7,16..23;
    //      lanes>=16 hold K 8..15,24..31; row = lane%16 ----
    union { unsigned u32[8]; v16bf v; } fa;
    const unsigned* lAu = (const unsigned*)bufA;
    #pragma unroll
    for (int j = 0; j < 4; ++j) {
      fa.u32[j]     = lAu[r * 16 + kb + j];
      fa.u32[4 + j] = lAu[r * 16 + kb + 8 + j];
    }

    // ---- NT B fragments + WMMAs, reusing the single A fragment ----
    #pragma unroll
    for (int s = 0; s < NT; ++s) {
      union { unsigned short u16[16]; v16bf v; } fb;
      #pragma unroll
      for (int i2 = 0; i2 < 16; ++i2)
        fb.u16[i2] = bufB[(kb2 + (i2 & 7) + ((i2 >> 3) << 4)) * (16 * NT) + s * 16 + nn];
      acc[s] = __builtin_amdgcn_wmma_f32_16x16x32_bf16(
          false, fa.v, false, fb.v, (short)0, acc[s], false, false);
    }

    okCur = okNext;
  }

  // ---- epilogue: C/D layout lane l -> col (l%16); VGPR j -> row j (+8 hi) ---
  const int rOff = (lane & 16) ? 8 : 0;
  #pragma unroll
  for (int s = 0; s < NT; ++s) {
    const int n = n0 + s * 16 + (lane & 15);
    float sc = 1.0f, sh = 0.0f;
    if (bn_g) {
      const float is = rsqrtf(bn_v[n] + 1e-5f);
      sc = bn_g[n] * is;
      sh = bn_b[n] - bn_m[n] * sc;
    }
    #pragma unroll
    for (int j = 0; j < 8; ++j) {
      const int grow = mBase + rOff + j;
      float val = acc[s][j];
      if (bn_g)  val = val * sc + sh;
      if (mask)  val *= mask[grow];
      if (resid) val += bf2f(resid[((long long)grow << csoS) + n]);
      if (relu)  val = fmaxf(val, 0.0f);
      out[((long long)grow << csoS) + n] = f2bf(val);
    }
  }
}

// ---------------------------------------------------------------------------
// Scatter voxel features + active mask into dense grid (channels-last, Cs=32).
// ---------------------------------------------------------------------------
__global__ void scatter_k(const float* __restrict__ feat,
                          const int* __restrict__ coords,
                          unsigned short* __restrict__ dense,
                          float* __restrict__ mask, int NV)
{
  const int i = blockIdx.x * 256 + threadIdx.x;
  if (i >= NV) return;
  const int b = coords[4 * i + 0], z = coords[4 * i + 1];
  const int y = coords[4 * i + 2], x = coords[4 * i + 3];
  const long long vox = (long long)((b * 64 + z) * 64 + y) * 64 + x;
  mask[vox] = 1.0f;
  #pragma unroll
  for (int c = 0; c < 4; ++c) dense[vox * 32 + c] = f2bf(feat[4 * i + c]);
}

// ---------------------------------------------------------------------------
// Active-set dilation: 3^3 / stride-2 / pad-1 max pool of the mask.
// ---------------------------------------------------------------------------
__global__ void pool_mask_k(const float* __restrict__ mi, float* __restrict__ mo,
                            int Din, int Do)
{
  const int i = blockIdx.x * 256 + threadIdx.x;
  const int tot = 2 * Do * Do * Do;
  if (i >= tot) return;
  const int b = i / (Do * Do * Do);
  const int rem = i % (Do * Do * Do);
  const int z = rem / (Do * Do), y = (rem / Do) % Do, x = rem % Do;
  float mx = 0.0f;
  for (int dz = 0; dz < 3; ++dz) {
    const int zi = 2 * z + dz - 1; if ((unsigned)zi >= (unsigned)Din) continue;
    for (int dy = 0; dy < 3; ++dy) {
      const int yi = 2 * y + dy - 1; if ((unsigned)yi >= (unsigned)Din) continue;
      for (int dx = 0; dx < 3; ++dx) {
        const int xi = 2 * x + dx - 1; if ((unsigned)xi >= (unsigned)Din) continue;
        mx = fmaxf(mx, mi[((long long)(b * Din + zi) * Din + yi) * Din + xi]);
      }
    }
  }
  mo[i] = mx;
}

// ---------------------------------------------------------------------------
// Repack f32 weight [Cout, Cin, 3,3,3] -> bf16 [27, Csi(pad), Cout].
// ---------------------------------------------------------------------------
__global__ void wconv_k(const float* __restrict__ w, unsigned short* __restrict__ wb,
                        int Cin, int Csi, int Cout)
{
  const int i = blockIdx.x * 256 + threadIdx.x;
  const int tot = 27 * Csi * Cout;
  if (i >= tot) return;
  const int co = i % Cout;
  const int ci = (i / Cout) % Csi;
  const int t  = i / (Cout * Csi);
  const float v = (ci < Cin) ? w[((long long)co * Cin + ci) * 27 + t] : 0.0f;
  wb[i] = f2bf(v);
}

// ---------------------------------------------------------------------------
// Concat x4 | x5_up | x6_up (channels-last bf16, 128 ch each, 8^3 grid)
// into NCDHW f32 [2, 384, 8, 8, 8].
// ---------------------------------------------------------------------------
__global__ void concat_k(const unsigned short* __restrict__ x4,
                         const unsigned short* __restrict__ x5u,
                         const unsigned short* __restrict__ x6u,
                         float* __restrict__ out, int total)
{
  const int i = blockIdx.x * 256 + threadIdx.x;
  if (i >= total) return;
  const int sp = i % 512;
  const int c  = (i / 512) % 384;
  const int b  = i / (512 * 384);
  const long long row = (long long)b * 512 + sp;
  float v;
  if (c < 128)      v = bf2f(x4 [row * 128 + c]);
  else if (c < 256) v = bf2f(x5u[row * 128 + (c - 128)]);
  else              v = bf2f(x6u[row * 128 + (c - 256)]);
  out[i] = v;
}

// ---------------------------------------------------------------------------
// Host orchestration.
// Input flattening assumption: top-level dict insertion order
// (voxel_features, voxel_coords, batch_size, params), params flattened as a
// JAX pytree (dict keys alphabetical):
//   downs[0..4]{bn{b,g,m,v}, w}, stages[0..5][0..1]{bn1{b,g,m,v},
//   bn2{b,g,m,v}, w1, w2}, stem_bn{b,g,m,v}, stem_w, up5_w, up6a_w, up6b_w.
// ---------------------------------------------------------------------------
static inline int ilog2(int x) { int s = 0; while ((1 << s) < x) ++s; return s; }

extern "C" void kernel_launch(void* const* d_in, const int* in_sizes, int n_in,
                              void* d_out, int out_size, void* d_ws, size_t ws_size,
                              hipStream_t stream)
{
  (void)n_in;
  static const int CHh[6] = {16, 32, 64, 128, 128, 128};

  // ---- workspace cursor allocation (byte offsets, 256B aligned) ----
  size_t cur = 0;
  auto alloc = [&](size_t elems, size_t esz) -> size_t {
    size_t off = cur;
    cur = (cur + elems * esz + 255) & ~(size_t)255;
    return off;
  };
  size_t VOX[6];
  for (int k = 0; k < 6; ++k) { const long long d = 64 >> k; VOX[k] = (size_t)(2 * d * d * d); }

  const size_t A0 = alloc(VOX[0] * 32, 2);      // dense input / L0 tmp
  size_t X[6], T[6];
  X[0] = alloc(VOX[0] * 32, 2);  T[0] = A0;
  X[1] = alloc(VOX[1] * 32, 2);  T[1] = alloc(VOX[1] * 32, 2);
  X[2] = alloc(VOX[2] * 64, 2);  T[2] = alloc(VOX[2] * 64, 2);
  for (int k = 3; k < 6; ++k) { X[k] = alloc(VOX[k] * 128, 2); T[k] = alloc(VOX[k] * 128, 2); }
  const size_t U5  = alloc(VOX[3] * 128, 2);
  const size_t U6M = alloc(VOX[4] * 128, 2);
  const size_t U6  = alloc(VOX[3] * 128, 2);
  size_t M[6];
  for (int k = 0; k < 6; ++k) M[k] = alloc(VOX[k], 4);

  auto wsz = [&](int cin, int cout) -> size_t {
    const int cs = cin < 32 ? 32 : cin;
    return (size_t)27 * cs * cout;
  };
  const size_t stemW = alloc(wsz(4, 16), 2);
  size_t stgW[6][2][2];
  for (int s = 0; s < 6; ++s)
    for (int bl = 0; bl < 2; ++bl)
      for (int c2 = 0; c2 < 2; ++c2)
        stgW[s][bl][c2] = alloc(wsz(CHh[s], CHh[s]), 2);
  size_t dwW[5];
  for (int i = 0; i < 5; ++i) dwW[i] = alloc(wsz(CHh[i], CHh[i + 1]), 2);
  size_t upW[3];
  for (int i = 0; i < 3; ++i) upW[i] = alloc(wsz(128, 128), 2);

  auto u16p = [&](size_t o) { return (unsigned short*)((char*)d_ws + o); };
  auto f32p = [&](size_t o) { return (float*)((char*)d_ws + o); };
  auto P    = [&](int idx)  { return (const float*)d_in[idx]; };

  // param index bases (see flattening assumption above)
  const int DWS = 3;            // downs: 5 x {bn.b,bn.g,bn.m,bn.v,w}
  const int STG = 3 + 25;       // stages: 12 blocks x {bn1*4, bn2*4, w1, w2}
  const int STEMBN = STG + 120; // stem_bn {b,g,m,v}
  const int STEMW  = STEMBN + 4;
  const int UP5W = STEMW + 1, UP6AW = STEMW + 2, UP6BW = STEMW + 3;

  // ---- zero workspace (padded channels must stay zero; deterministic) ----
  (void)hipMemsetAsync(d_ws, 0, cur < ws_size ? cur : ws_size, stream);

  // ---- repack all weights to bf16 [27, Csi, Cout] ----
  auto wcv = [&](int pidx, size_t off, int cin, int cout) {
    const int cs = cin < 32 ? 32 : cin;
    const int tot = 27 * cs * cout;
    wconv_k<<<dim3((tot + 255) / 256), dim3(256), 0, stream>>>(
        (const float*)d_in[pidx], u16p(off), cin, cs, cout);
  };
  wcv(STEMW, stemW, 4, 16);
  for (int s = 0; s < 6; ++s)
    for (int bl = 0; bl < 2; ++bl) {
      const int base = STG + (s * 2 + bl) * 10;
      wcv(base + 8, stgW[s][bl][0], CHh[s], CHh[s]);
      wcv(base + 9, stgW[s][bl][1], CHh[s], CHh[s]);
    }
  for (int i = 0; i < 5; ++i) wcv(DWS + 5 * i + 4, dwW[i], CHh[i], CHh[i + 1]);
  wcv(UP5W, upW[0], 128, 128);
  wcv(UP6AW, upW[1], 128, 128);
  wcv(UP6BW, upW[2], 128, 128);

  // ---- scatter voxels + mask at L0 ----
  const int NV = in_sizes[0] / 4;
  scatter_k<<<dim3((NV + 255) / 256), dim3(256), 0, stream>>>(
      (const float*)d_in[0], (const int*)d_in[1], u16p(A0), f32p(M[0]), NV);

  // ---- conv launcher (NT = N sub-tiles per wave) ----
  auto conv = [&](unsigned short* inp, unsigned short* outp, unsigned short* w,
                  int bnIdx, float* mask, unsigned short* resid,
                  int Din, int Do, int Cin, int Cout, int mode, int relu) {
    const float *g = nullptr, *bb = nullptr, *mm = nullptr, *vv = nullptr;
    if (bnIdx >= 0) { bb = P(bnIdx); g = P(bnIdx + 1); mm = P(bnIdx + 2); vv = P(bnIdx + 3); }
    const int Csi = Cin < 32 ? 32 : Cin;
    const int Cso = Cout < 32 ? 32 : Cout;
    const int NT  = (Cout >= 64) ? 4 : (Cout == 32 ? 2 : 1);
    const int dinS = ilog2(Din), doS = ilog2(Do);
    const int csiS = ilog2(Csi), csoS = ilog2(Cso), coutS = ilog2(Cout);
    dim3 grid((unsigned)(2 * Do * Do * Do / 16), (unsigned)(Cout / (16 * NT)));
    switch (NT) {
      case 1:
        conv_wmma_k<1><<<grid, dim3(32), 0, stream>>>(
            inp, outp, w, g, bb, mm, vv, mask, resid,
            Din, dinS, doS, csiS, csoS, coutS, mode, relu);
        break;
      case 2:
        conv_wmma_k<2><<<grid, dim3(32), 0, stream>>>(
            inp, outp, w, g, bb, mm, vv, mask, resid,
            Din, dinS, doS, csiS, csoS, coutS, mode, relu);
        break;
      default:
        conv_wmma_k<4><<<grid, dim3(32), 0, stream>>>(
            inp, outp, w, g, bb, mm, vv, mask, resid,
            Din, dinS, doS, csiS, csoS, coutS, mode, relu);
        break;
    }
  };

  // ---- stem: relu(bn(conv(dense)) * m0) ----
  conv(u16p(A0), u16p(X[0]), u16p(stemW), STEMBN, f32p(M[0]), nullptr,
       64, 64, 4, 16, 0, 1);

  // ---- stage 0 residual blocks (in-place residual) ----
  for (int bl = 0; bl < 2; ++bl) {
    const int base = STG + bl * 10;
    conv(u16p(X[0]), u16p(T[0]), u16p(stgW[0][bl][0]), base, f32p(M[0]), nullptr,
         64, 64, 16, 16, 0, 1);
    conv(u16p(T[0]), u16p(X[0]), u16p(stgW[0][bl][1]), base + 4, f32p(M[0]), u16p(X[0]),
         64, 64, 16, 16, 0, 1);
  }

  // ---- down i (stride-2) + stage i+1 blocks ----
  for (int i = 0; i < 5; ++i) {
    const int Din = 64 >> i, Do = 64 >> (i + 1);
    const int tot = 2 * Do * Do * Do;
    pool_mask_k<<<dim3((tot + 255) / 256), dim3(256), 0, stream>>>(
        f32p(M[i]), f32p(M[i + 1]), Din, Do);
    conv(u16p(X[i]), u16p(X[i + 1]), u16p(dwW[i]), DWS + 5 * i, f32p(M[i + 1]), nullptr,
         Din, Do, CHh[i], CHh[i + 1], 1, 1);
    const int c = CHh[i + 1];
    for (int bl = 0; bl < 2; ++bl) {
      const int base = STG + ((i + 1) * 2 + bl) * 10;
      conv(u16p(X[i + 1]), u16p(T[i + 1]), u16p(stgW[i + 1][bl][0]), base,
           f32p(M[i + 1]), nullptr, Do, Do, c, c, 0, 1);
      conv(u16p(T[i + 1]), u16p(X[i + 1]), u16p(stgW[i + 1][bl][1]), base + 4,
           f32p(M[i + 1]), u16p(X[i + 1]), Do, Do, c, c, 0, 1);
    }
  }

  // ---- upsample paths (transposed stride-2 conv, mask only) ----
  conv(u16p(X[4]), u16p(U5),  u16p(upW[0]), -1, f32p(M[3]), nullptr, 4, 8, 128, 128, 2, 0);
  conv(u16p(X[5]), u16p(U6M), u16p(upW[1]), -1, f32p(M[4]), nullptr, 2, 4, 128, 128, 2, 0);
  conv(u16p(U6M),  u16p(U6),  u16p(upW[2]), -1, f32p(M[3]), nullptr, 4, 8, 128, 128, 2, 0);

  // ---- concat to NCDHW f32 output ----
  concat_k<<<dim3((out_size + 255) / 256), dim3(256), 0, stream>>>(
      u16p(X[3]), u16p(U5), u16p(U6), (float*)d_out, out_size);
}
